// GNN_Agent_37074157699336
// MI455X (gfx1250) — matrix operand order repaired and checked by reading, not compiled, not run on hardware
//
#include <hip/hip_runtime.h>
#include <hip/hip_bf16.h>
#include <math.h>

typedef __attribute__((ext_vector_type(16))) _Float16 v16h;
typedef __attribute__((ext_vector_type(8)))  float    v8f;

#define NNODES 20000
#define NEDGES 320000
#define BATCH  8
#define HDIM   64
#define FEAT   63
#define COVL   10000
#define HIST   64
#define MLPH   2048
#define CIN    (NNODES + HIST)          // 20064
#define ROWS   (BATCH * NNODES)         // 160000
#define XROW   (COVL + HIST)            // 10064
#define KCH    (CIN / 32)               // 627 K-chunks for graph MLP

// ---------------- init: h[b,n,0:63]=nodes[n,:], h[b,n,63]=0 ----------------
__global__ void k_init_h(float* __restrict__ h, const float* __restrict__ nodes) {
    size_t idx = (size_t)blockIdx.x * 256 + threadIdx.x;   // < 10,240,000
    int f = (int)(idx & 63);
    size_t row = idx >> 6;
    int n = (int)(row % NNODES);
    h[idx] = (f < FEAT) ? nodes[(size_t)n * FEAT + f] : 0.0f;
}

// scatter cov into h[...,63] at count2label positions (indices are unique)
__global__ void k_scatter_cov(float* __restrict__ h, const float* __restrict__ x,
                              const int* __restrict__ c2l) {
    int t = blockIdx.x * 256 + threadIdx.x;
    if (t >= BATCH * COVL) return;
    int b = t / COVL, i = t % COVL;
    int lbl = c2l[i];
    h[((size_t)b * NNODES + lbl) * HDIM + (HDIM - 1)] = x[(size_t)b * XROW + i];
}

// concat[b, 20000+j] = history
__global__ void k_history(float* __restrict__ cc, const float* __restrict__ x) {
    int t = blockIdx.x * 256 + threadIdx.x;
    if (t >= BATCH * HIST) return;
    int b = t >> 6, j = t & 63;
    cc[(size_t)b * CIN + NNODES + j] = x[(size_t)b * XROW + COVL + j];
}

// ---------------- conv: m = h @ W  (rows=160000, 64x64), WMMA f16 ----------
__global__ void k_conv(const float* __restrict__ h, const float* __restrict__ W,
                       float* __restrict__ m) {
    int wave = threadIdx.x >> 5;            // 0..7
    int lane = threadIdx.x & 31;
    int l16  = lane & 15;
    int hi   = lane >> 4;                   // 0 or 1
    int rowBase = blockIdx.x * 32 + (wave >> 2) * 16;
    int colBase = (wave & 3) * 16;
    const float* arow = h + (size_t)(rowBase + l16) * HDIM;
    v8f c = {};
    #pragma unroll
    for (int kc = 0; kc < 2; ++kc) {
        int kbase = kc * 32;
        int ka = kbase + hi * 16;           // A: K = ka..ka+15
        v16h a, b;
        #pragma unroll
        for (int e = 0; e < 8; ++e) {
            a[e]     = (_Float16)arow[ka + e];
            a[e + 8] = (_Float16)arow[ka + 8 + e];
        }
        int kb = kbase + hi * 16;           // B: K = kb..kb+15, N fixed
        #pragma unroll
        for (int e = 0; e < 16; ++e)
            b[e] = (_Float16)W[(size_t)(kb + e) * HDIM + colBase + l16];
        c = __builtin_amdgcn_wmma_f32_16x16x32_f16(false, a, false, b,
                                                   (short)0, c, false, false);
    }
    #pragma unroll
    for (int v = 0; v < 8; ++v)
        m[(size_t)(rowBase + v + hi * 8) * HDIM + colBase + l16] = c[v];
}

// ---------------- zero agg (float4) ----------------------------------------
__global__ void k_zero(float4* __restrict__ p) {
    p[(size_t)blockIdx.x * 256 + threadIdx.x] = make_float4(0.f, 0.f, 0.f, 0.f);
}

// ---------------- edge gather + scatter-add: agg[:,dst,:] += m[:,src,:] ----
// 2 edges per block; 128 threads per edge; float4 gather + 4x f32 atomic add
__global__ void k_edge(const float* __restrict__ m, float* __restrict__ agg,
                       const int* __restrict__ esrc, const int* __restrict__ edst) {
    int tid = threadIdx.x;
    int e = blockIdx.x * 2 + (tid >> 7);
    int q = tid & 127;                      // (b, f/4) slot
    int b = q >> 4;
    int f4 = (q & 15) * 4;
    int s = esrc[e], d = edst[e];
    const float4 v = *(const float4*)(m + ((size_t)b * NNODES + s) * HDIM + f4);
    float* dst = agg + ((size_t)b * NNODES + d) * HDIM + f4;
    atomicAdd(dst + 0, v.x);
    atomicAdd(dst + 1, v.y);
    atomicAdd(dst + 2, v.z);
    atomicAdd(dst + 3, v.w);
}

// ---------------- GRU cell: h = GRU(agg, h); WMMA for gi/gh ----------------
__global__ void k_gru(float* __restrict__ h, const float* __restrict__ agg,
                      const float* __restrict__ wih, const float* __restrict__ whh,
                      const float* __restrict__ bih, const float* __restrict__ bhh) {
    __shared__ float gs[2][16][196];        // [gi|gh][row][0..191], padded
    int wave = threadIdx.x >> 5;
    int lane = threadIdx.x & 31;
    int l16  = lane & 15;
    int hi   = lane >> 4;
    int rowBase = blockIdx.x * 16;
    #pragma unroll
    for (int jj = 0; jj < 3; ++jj) {
        int job = wave * 3 + jj;            // 0..23
        int mat = job / 12;                 // 0: gi=agg@wih^T, 1: gh=h@whh^T
        int ct  = job % 12;
        int colBase = ct * 16;
        const float* A  = (mat == 0 ? agg : h) + (size_t)(rowBase + l16) * HDIM;
        const float* Wt = (mat == 0 ? wih : whh);
        v8f c = {};
        #pragma unroll
        for (int kc = 0; kc < 2; ++kc) {
            int kbase = kc * 32;
            int ka = kbase + hi * 16;
            v16h a, b;
            #pragma unroll
            for (int e = 0; e < 8; ++e) {
                a[e]     = (_Float16)A[ka + e];
                a[e + 8] = (_Float16)A[ka + 8 + e];
            }
            int kb = kbase + hi * 16;
            const float* wrow = Wt + (size_t)(colBase + l16) * HDIM + kb; // contiguous
            #pragma unroll
            for (int e = 0; e < 16; ++e) b[e] = (_Float16)wrow[e];
            c = __builtin_amdgcn_wmma_f32_16x16x32_f16(false, a, false, b,
                                                       (short)0, c, false, false);
        }
        #pragma unroll
        for (int v = 0; v < 8; ++v)
            gs[mat][v + hi * 8][colBase + l16] = c[v];
    }
    __syncthreads();
    // gates: 16 rows x 64 cols
    for (int t = threadIdx.x; t < 16 * 64; t += 256) {
        int r = t >> 6, j = t & 63;
        float ir = gs[0][r][j]        + bih[j];
        float iz = gs[0][r][64 + j]   + bih[64 + j];
        float inn= gs[0][r][128 + j]  + bih[128 + j];
        float hr = gs[1][r][j]        + bhh[j];
        float hz = gs[1][r][64 + j]   + bhh[64 + j];
        float hn = gs[1][r][128 + j]  + bhh[128 + j];
        float rg = 1.0f / (1.0f + __expf(-(ir + hr)));
        float z  = 1.0f / (1.0f + __expf(-(iz + hz)));
        float nn = tanhf(inn + rg * hn);
        size_t idx = (size_t)(rowBase + r) * HDIM + j;
        float hold = h[idx];
        h[idx] = (1.0f - z) * nn + z * hold;
    }
}

// ---------------- node MLP: concat[b, n] = relu(h[b,n,:].w + b) ------------
__global__ void k_nodemlp(const float* __restrict__ h, const float* __restrict__ w,
                          const float* __restrict__ bias, float* __restrict__ cc) {
    int r = blockIdx.x * 256 + threadIdx.x;   // < 160000
    int b = r / NNODES, n = r % NNODES;
    const float4* hv = (const float4*)(h + (size_t)r * HDIM);
    const float4* wv = (const float4*)w;
    float s = 0.f;
    #pragma unroll
    for (int i = 0; i < 16; ++i) {
        float4 a = hv[i], q = wv[i];
        s += a.x * q.x + a.y * q.y + a.z * q.z + a.w * q.w;
    }
    s += bias[0];
    cc[(size_t)b * CIN + n] = s > 0.f ? s : 0.f;
}

// ---------------- graph MLP: state = relu(concat @ gw^T + gb), WMMA --------
// Block: 512 threads = 16 waves, one 16-col tile of the 2048 outputs.
// M=16 (batch rows 0..7, rows 8..15 duplicated & discarded), K=20064 split
// across the 16 waves (chunk stride 16), partials reduced through LDS.
__global__ void __launch_bounds__(512)
k_graphmlp(const float* __restrict__ cc, const float* __restrict__ w,
           const float* __restrict__ bias, float* __restrict__ st) {
    __shared__ float part[16][16][20];      // [wave][M][N], padded (20 KB)
    int tid  = threadIdx.x;
    int wave = tid >> 5;                    // 0..15
    int lane = tid & 31;
    int l16  = lane & 15;
    int hi   = lane >> 4;
    int colBase = blockIdx.x * 16;
    int br = l16 & 7;                       // clamp A row into valid batch range
    const float* arow = cc + (size_t)br * CIN;
    const float* wrow = w + (size_t)(colBase + l16) * CIN;
    v8f c = {};
    for (int ch = wave; ch < KCH; ch += 16) {
        int kbase = ch * 32;
        int ka = kbase + hi * 16;
        int kb = kbase + hi * 16;
        v16h a, b;
        #pragma unroll
        for (int e = 0; e < 8; ++e) {
            a[e]     = (_Float16)arow[ka + e];
            a[e + 8] = (_Float16)arow[ka + 8 + e];
        }
        #pragma unroll
        for (int e = 0; e < 16; ++e) b[e] = (_Float16)wrow[kb + e];
        c = __builtin_amdgcn_wmma_f32_16x16x32_f16(false, a, false, b,
                                                   (short)0, c, false, false);
    }
    #pragma unroll
    for (int v = 0; v < 8; ++v)
        part[wave][v + hi * 8][l16] = c[v];
    __syncthreads();
    if (tid < 256) {
        int r = tid >> 4, n = tid & 15;
        float s = 0.f;
        #pragma unroll
        for (int wv = 0; wv < 16; ++wv) s += part[wv][r][n];
        if (r < 8) {
            float v = s + bias[colBase + n];
            st[(size_t)r * MLPH + colBase + n] = v > 0.f ? v : 0.f;
        }
    }
}

// ---------------- heads: out[b, 0:64]=actor, out[b,64]=critic --------------
__global__ void k_head(const float* __restrict__ st,
                       const float* __restrict__ aw, const float* __restrict__ ab,
                       const float* __restrict__ cw, const float* __restrict__ cb,
                       float* __restrict__ out) {
    __shared__ float red[8][256];
    int o = blockIdx.x;                     // 0..64
    int tid = threadIdx.x;
    const float* wv = (o < HIST) ? (aw + (size_t)o * MLPH) : cw;
    float bs = (o < HIST) ? ab[o] : cb[0];
    float acc[8] = {};
    for (int k = tid; k < MLPH; k += 256) {
        float w = wv[k];
        #pragma unroll
        for (int b = 0; b < 8; ++b) acc[b] += w * st[(size_t)b * MLPH + k];
    }
    #pragma unroll
    for (int b = 0; b < 8; ++b) red[b][tid] = acc[b];
    __syncthreads();
    for (int s2 = 128; s2 > 0; s2 >>= 1) {
        if (tid < s2) {
            #pragma unroll
            for (int b = 0; b < 8; ++b) red[b][tid] += red[b][tid + s2];
        }
        __syncthreads();
    }
    if (tid < 8) out[(size_t)tid * (HIST + 1) + o] = red[tid][0] + bs;
}

extern "C" void kernel_launch(void* const* d_in, const int* in_sizes, int n_in,
                              void* d_out, int out_size, void* d_ws, size_t ws_size,
                              hipStream_t stream) {
    const float* x     = (const float*)d_in[0];
    const float* nodes = (const float*)d_in[1];
    const float* convw = (const float*)d_in[2];
    const float* wih   = (const float*)d_in[3];
    const float* whh   = (const float*)d_in[4];
    const float* bih   = (const float*)d_in[5];
    const float* bhh   = (const float*)d_in[6];
    const float* nmw   = (const float*)d_in[7];
    const float* nmb   = (const float*)d_in[8];
    const float* gw    = (const float*)d_in[9];
    const float* gb    = (const float*)d_in[10];
    const float* aw    = (const float*)d_in[11];
    const float* ab    = (const float*)d_in[12];
    const float* cw    = (const float*)d_in[13];
    const float* cb    = (const float*)d_in[14];
    const int*   c2l   = (const int*)d_in[15];
    const int*   edges = (const int*)d_in[16];
    const int*   esrc  = edges;
    const int*   edst  = edges + NEDGES;

    char* ws = (char*)d_ws;
    const size_t hbytes = (size_t)ROWS * HDIM * sizeof(float);     // 40,960,000
    float* h   = (float*)(ws);
    float* m   = (float*)(ws + hbytes);
    float* agg = (float*)(ws + 2 * hbytes);
    float* cc  = (float*)(ws + 3 * hbytes);
    float* st  = (float*)(ws + 3 * hbytes + (size_t)BATCH * CIN * sizeof(float));

    k_init_h<<<(ROWS * HDIM) / 256, 256, 0, stream>>>(h, nodes);
    k_scatter_cov<<<(BATCH * COVL + 255) / 256, 256, 0, stream>>>(h, x, c2l);
    k_history<<<2, 256, 0, stream>>>(cc, x);

    for (int l = 0; l < 3; ++l) {
        k_conv<<<ROWS / 32, 256, 0, stream>>>(h, convw + (size_t)l * HDIM * HDIM, m);
        k_zero<<<(ROWS * HDIM) / (4 * 256), 256, 0, stream>>>((float4*)agg);
        k_edge<<<NEDGES / 2, 256, 0, stream>>>(m, agg, esrc, edst);
        k_gru<<<ROWS / 16, 256, 0, stream>>>(h, agg, wih, whh, bih, bhh);
    }

    k_nodemlp<<<ROWS / 256, 256, 0, stream>>>(h, nmw, nmb, cc);
    k_graphmlp<<<MLPH / 16, 512, 0, stream>>>(cc, gw, gb, st);
    k_head<<<HIST + 1, 256, 0, stream>>>(st, aw, ab, cw, cb, (float*)d_out);
}
